// QKVAttention_3942779977839
// MI455X (gfx1250) — compile-verified, matrix operand order checked by reading, not run on hardware
//
#include <hip/hip_runtime.h>
#include <math.h>

typedef __attribute__((ext_vector_type(16))) _Float16 v16h;
typedef __attribute__((ext_vector_type(8)))  _Float16 v8h;
typedef __attribute__((ext_vector_type(8)))  float    v8f;
typedef __attribute__((ext_vector_type(4)))  float    v4f;

#define T_DIM   2048
#define C_DIM   64
#define TM      128              // t rows per workgroup
#define SCHUNK  32               // s columns per inner iteration
#define NCHUNK  (T_DIM / SCHUNK) // 64
#define NWAVES  8

// Load a 16-bit WMMA fragment (A 16x32 rows=m, or B 32x16 cols=n) from LDS.
// gfx1250 layout: lane L -> row/col (L&15); elements e=0..7 -> k = e + khi,
// e=8..15 -> k = 16 + (e-8) + khi, where khi = (L>=16)*8.
__device__ inline v16h load_frag(const _Float16* base, int stride_h, int lane) {
    const int row = lane & 15;
    const int khi = (lane >> 4) << 3;
    const _Float16* p = base + row * stride_h + khi;
    v8h lo = *(const v8h*)(p);
    v8h hi = *(const v8h*)(p + 16);
    return __builtin_shufflevector(lo, hi, 0,1,2,3,4,5,6,7,8,9,10,11,12,13,14,15);
}

// Exchange with lane^16 via v_permlanex16 (identity selects) — pure VALU.
__device__ inline float xchg16(float x) {
    int xi = __builtin_bit_cast(int, x);
    int yi = __builtin_amdgcn_permlanex16(xi, xi, 0x76543210, 0xfedcba98, false, false);
    return __builtin_bit_cast(float, yi);
}

__global__ __launch_bounds__(256)
void qkv_attn_kernel(const float* __restrict__ qkv, float* __restrict__ out) {
    __shared__ __align__(16) _Float16 Qt[TM * C_DIM];        // [t][c]          16KB
    __shared__ __align__(16) _Float16 Kt[2][SCHUNK * C_DIM]; // [buf][s][c]      8KB
    __shared__ __align__(16) _Float16 Vl[2][C_DIM * SCHUNK]; // [buf][c][s]      8KB

    const int tid  = threadIdx.x;
    const int lane = tid & 31;
    const int wave = tid >> 5;
    const int bh    = blockIdx.x >> 4;   // 16 t-tiles per head
    const int ttile = blockIdx.x & 15;
    const int t0    = ttile * TM;
    const int b = bh >> 3, h = bh & 7;

    const float scale = 0.02209708691207961f; // 1/sqrt(2048), applied to Q and K

    const float* Qg = qkv + (size_t)(b * 1536 +        h * 64) * T_DIM;
    const float* Kg = qkv + (size_t)(b * 1536 +  512 + h * 64) * T_DIM;
    const float* Vg = qkv + (size_t)(b * 1536 + 1024 + h * 64) * T_DIM;

    // Per-thread K/V staging coordinates: two strips j = tid, tid+256 over
    // 64(c) x 8(s4) float4 tiles per chunk.
    const int s4a = (tid)       & 7, ca = (tid)       >> 3;
    const int s4b = (tid + 256) & 7, cb = (tid + 256) >> 3;

    // ---- Prologue: stage chunk 0 K/V + full Q tile ----
    {
        v4f k0 = *(const v4f*)(Kg + (size_t)ca * T_DIM + s4a * 4);
        v4f v0 = *(const v4f*)(Vg + (size_t)ca * T_DIM + s4a * 4);
        v4f k1 = *(const v4f*)(Kg + (size_t)cb * T_DIM + s4b * 4);
        v4f v1 = *(const v4f*)(Vg + (size_t)cb * T_DIM + s4b * 4);
        #pragma unroll
        for (int k = 0; k < 4; ++k) {
            Kt[0][(s4a * 4 + k) * C_DIM + ca] = (_Float16)(k0[k] * scale);
            Vl[0][ca * SCHUNK + s4a * 4 + k]  = (_Float16)v0[k];
            Kt[0][(s4b * 4 + k) * C_DIM + cb] = (_Float16)(k1[k] * scale);
            Vl[0][cb * SCHUNK + s4b * 4 + k]  = (_Float16)v1[k];
        }
    }
    for (int j = tid; j < (TM / 4) * C_DIM; j += 256) {
        int t4 = j & 31;
        int c  = j >> 5;
        v4f q = *(const v4f*)(Qg + (size_t)c * T_DIM + t0 + t4 * 4);
        #pragma unroll
        for (int k = 0; k < 4; ++k)
            Qt[(t4 * 4 + k) * C_DIM + c] = (_Float16)(q[k] * scale);
    }
    __syncthreads();

    // Per-wave B fragments of Q (cols n = t = wave*16 + (lane&15)), split over k=c.
    const _Float16* qbase = Qt + (wave * 16) * C_DIM;
    v16h bq0 = load_frag(qbase,      C_DIM, lane);   // c = 0..31
    v16h bq1 = load_frag(qbase + 32, C_DIM, lane);   // c = 32..63

    // Online-softmax state: lane owns row t = t0 + wave*16 + (lane&15);
    // both lane halves keep identical full-row stats.
    float mst = -__builtin_inff();
    float lst = 0.0f;
    v8f acc[4];   // D[m=c][n=t]: lane n=t, vgpr r -> c = ct*16 + r + 8*(lane>=16)
    #pragma unroll
    for (int ct = 0; ct < 4; ++ct) acc[ct] = (v8f){0,0,0,0,0,0,0,0};

    v4f kra, vra, krb, vrb;

    for (int it = 0; it < NCHUNK; ++it) {
        const _Float16* kb = Kt[it & 1];
        const _Float16* vb = Vl[it & 1];
        _Float16* kn = Kt[(it & 1) ^ 1];
        _Float16* vn = Vl[(it & 1) ^ 1];
        const bool more = (it + 1 < NCHUNK);

        // ---- Issue next chunk's global loads early (overlap with compute) ----
        if (more) {
            const int s0n = (it + 1) * SCHUNK;
            kra = *(const v4f*)(Kg + (size_t)ca * T_DIM + s0n + s4a * 4);
            vra = *(const v4f*)(Vg + (size_t)ca * T_DIM + s0n + s4a * 4);
            krb = *(const v4f*)(Kg + (size_t)cb * T_DIM + s0n + s4b * 4);
            vrb = *(const v4f*)(Vg + (size_t)cb * T_DIM + s0n + s4b * 4);
        }

        // ---- Issue all LDS fragment loads up front (partial dscnt waits) ----
        v16h ak0 = load_frag(kb,                   C_DIM, lane);
        v16h ak1 = load_frag(kb + 32,              C_DIM, lane);
        v16h ak2 = load_frag(kb + 16 * C_DIM,      C_DIM, lane);
        v16h ak3 = load_frag(kb + 16 * C_DIM + 32, C_DIM, lane);
        v16h av0 = load_frag(vb + 0 * 16 * SCHUNK, SCHUNK, lane);
        v16h av1 = load_frag(vb + 1 * 16 * SCHUNK, SCHUNK, lane);
        v16h av2 = load_frag(vb + 2 * 16 * SCHUNK, SCHUNK, lane);
        v16h av3 = load_frag(vb + 3 * 16 * SCHUNK, SCHUNK, lane);

        // ---- Transposed scores: D[m=s][n=t], two 16-s tiles, K-dim = 64 ----
        v8f sc0 = (v8f){0,0,0,0,0,0,0,0};
        sc0 = __builtin_amdgcn_wmma_f32_16x16x32_f16(false, ak0, false, bq0,
                                                     (short)0, sc0, false, false);
        sc0 = __builtin_amdgcn_wmma_f32_16x16x32_f16(false, ak1, false, bq1,
                                                     (short)0, sc0, false, false);
        v8f sc1 = (v8f){0,0,0,0,0,0,0,0};
        sc1 = __builtin_amdgcn_wmma_f32_16x16x32_f16(false, ak2, false, bq0,
                                                     (short)0, sc1, false, false);
        sc1 = __builtin_amdgcn_wmma_f32_16x16x32_f16(false, ak3, false, bq1,
                                                     (short)0, sc1, false, false);

        // ---- Online softmax over the 32 s-values of this lane's row t ----
        float cm = fmaxf(sc0[0], sc1[0]);
        #pragma unroll
        for (int e = 1; e < 8; ++e) cm = fmaxf(cm, fmaxf(sc0[e], sc1[e]));
        cm = fmaxf(cm, xchg16(cm));            // merge partner half's s-range
        float mnew  = fmaxf(mst, cm);
        float alpha = __expf(mst - mnew);
        mst = mnew;

        float p0[8], p1[8];
        float ls = 0.0f;
        #pragma unroll
        for (int e = 0; e < 8; ++e) {
            p0[e] = __expf(sc0[e] - mnew);
            p1[e] = __expf(sc1[e] - mnew);
            ls += p0[e] + p1[e];
        }
        ls += xchg16(ls);
        lst = lst * alpha + ls;

        // P^T already in B-fragment layout: e<8 -> s=e+khi (tile0),
        // e>=8 -> s=16+(e-8)+khi (tile1). Convert to f16.
        v16h pf;
        #pragma unroll
        for (int e = 0; e < 8; ++e) {
            pf[e]     = (_Float16)p0[e];
            pf[e + 8] = (_Float16)p1[e];
        }

        // ---- Rescale accumulators (alpha uniform per lane = per row t) ----
        #pragma unroll
        for (int ct = 0; ct < 4; ++ct) acc[ct] *= alpha;

        // ---- PV: D[m=c][n=t] += V(16x32) x P^T(32x16) for 4 c-tiles ----
        acc[0] = __builtin_amdgcn_wmma_f32_16x16x32_f16(false, av0, false, pf,
                                                        (short)0, acc[0], false, false);
        acc[1] = __builtin_amdgcn_wmma_f32_16x16x32_f16(false, av1, false, pf,
                                                        (short)0, acc[1], false, false);
        acc[2] = __builtin_amdgcn_wmma_f32_16x16x32_f16(false, av2, false, pf,
                                                        (short)0, acc[2], false, false);
        acc[3] = __builtin_amdgcn_wmma_f32_16x16x32_f16(false, av3, false, pf,
                                                        (short)0, acc[3], false, false);

        // ---- Convert + store the prefetched chunk into the other buffer ----
        if (more) {
            #pragma unroll
            for (int k = 0; k < 4; ++k) {
                kn[(s4a * 4 + k) * C_DIM + ca] = (_Float16)(kra[k] * scale);
                vn[ca * SCHUNK + s4a * 4 + k]  = (_Float16)vra[k];
                kn[(s4b * 4 + k) * C_DIM + cb] = (_Float16)(krb[k] * scale);
                vn[cb * SCHUNK + s4b * 4 + k]  = (_Float16)vrb[k];
            }
        }
        __syncthreads();
    }

    // ---- Epilogue: divide by row sum, store out[b][h*64+c][t] ----
    const size_t obase = (size_t)(b * 512 + h * 64) * T_DIM;
    const float inv = 1.0f / lst;
    const int tl = t0 + wave * 16 + (lane & 15);
    #pragma unroll
    for (int ct = 0; ct < 4; ++ct) {
        #pragma unroll
        for (int r = 0; r < 8; ++r) {
            int c = ct * 16 + r + ((lane >> 4) << 3);
            out[obase + (size_t)c * T_DIM + tl] = acc[ct][r] * inv;
        }
    }
}

extern "C" void kernel_launch(void* const* d_in, const int* in_sizes, int n_in,
                              void* d_out, int out_size, void* d_ws, size_t ws_size,
                              hipStream_t stream) {
    (void)in_sizes; (void)n_in; (void)d_ws; (void)ws_size; (void)out_size;
    const float* qkv = (const float*)d_in[0];
    float* out = (float*)d_out;
    // grid = 32 batch-heads * 16 t-tiles of 128 rows
    qkv_attn_kernel<<<dim3(32 * 16), dim3(256), 0, stream>>>(qkv, out);
}
